// RobustImageClassifier_10247791968454
// MI455X (gfx1250) — compile-verified, matrix operand order checked by reading, not run on hardware
//
#include <hip/hip_runtime.h>
#include <hip/hip_bf16.h>
#include <stdint.h>

// ---------------------------------------------------------------------------
// CDNA5 (gfx1250): bf16 WMMA FC GEMM (LDS-staged A via async global->LDS,
// in-block split-K), WMMA pairwise Gram, VALU/LDS OT mirror-descent solver.
// ---------------------------------------------------------------------------

typedef __bf16 v16bf __attribute__((ext_vector_type(16)));
typedef float  v8f   __attribute__((ext_vector_type(8)));
typedef int    v4i   __attribute__((ext_vector_type(4)));

union Frag16 { uint4 q[2]; unsigned int u[8]; v16bf v; };

#if defined(__has_builtin)
#  if __has_builtin(__builtin_amdgcn_global_load_async_to_lds_b128) && \
      __has_builtin(__builtin_amdgcn_s_wait_asynccnt)
#    define USE_ASYNC_LDS 1
#  endif
#endif
#ifndef USE_ASYNC_LDS
#  define USE_ASYNC_LDS 0
#endif

__device__ __forceinline__ void stage16(const unsigned short* g, unsigned short* l) {
#if USE_ASYNC_LDS
  __builtin_amdgcn_global_load_async_to_lds_b128(
      (__attribute__((address_space(1))) v4i*)g,
      (__attribute__((address_space(3))) v4i*)l, 0, 0);
#else
  *(uint4*)l = *(const uint4*)g;
#endif
}
__device__ __forceinline__ void stage_wait() {
#if USE_ASYNC_LDS
  __builtin_amdgcn_s_wait_asynccnt(0);
#endif
}

__device__ __forceinline__ unsigned short f2bf(float f) {
  unsigned int u = __float_as_uint(f);
  u += 0x7FFFu + ((u >> 16) & 1u);           // round-to-nearest-even
  return (unsigned short)(u >> 16);
}

// ---------------------------------------------------------------------------
// 1) conv3x3 (3->32ch, VALID) + ReLU + 2x2 maxpool -> bf16 [1536, 30752]
// ---------------------------------------------------------------------------
__global__ void __launch_bounds__(256)
conv_pool_kernel(const float* __restrict__ X, const float* __restrict__ W,
                 const float* __restrict__ B, unsigned short* __restrict__ pooled) {
  __shared__ float wsh[864];
  __shared__ float bsh[32];
  for (int t = threadIdx.x; t < 864; t += 256) wsh[t] = W[t];
  for (int t = threadIdx.x; t < 32;  t += 256) bsh[t] = B[t];
  __syncthreads();

  long long idx = (long long)blockIdx.x * 256 + threadIdx.x;
  if (idx >= 47235072LL) return;               // 1536 * 30752
  int n  = (int)(idx / 30752);
  int r  = (int)(idx % 30752);
  int c  = r / 961;
  int r2 = r % 961;
  int ph = r2 / 31, pw = r2 % 31;

  const float* xn = X + (long long)n * 12288;
  const float* wc = wsh + c * 27;

  float best = 0.0f;
  #pragma unroll
  for (int dy = 0; dy < 2; ++dy) {
    #pragma unroll
    for (int dx = 0; dx < 2; ++dx) {
      int oh = 2 * ph + dy, ow = 2 * pw + dx;
      float acc = bsh[c];
      #pragma unroll
      for (int ic = 0; ic < 3; ++ic)
        #pragma unroll
        for (int ky = 0; ky < 3; ++ky)
          #pragma unroll
          for (int kx = 0; kx < 3; ++kx)
            acc = fmaf(xn[ic * 4096 + (oh + ky) * 64 + (ow + kx)],
                       wc[ic * 9 + ky * 3 + kx], acc);
      acc  = fmaxf(acc, 0.0f);
      best = fmaxf(best, acc);
    }
  }
  pooled[idx] = f2bf(best);
}

// ---------------------------------------------------------------------------
// 2) fc_w fp32 -> bf16
// ---------------------------------------------------------------------------
__global__ void __launch_bounds__(256)
cast_w_kernel(const float* __restrict__ src, unsigned short* __restrict__ dst, int n) {
  int idx = blockIdx.x * 256 + threadIdx.x;
  if (idx < n) dst[idx] = f2bf(src[idx]);
}

// ---------------------------------------------------------------------------
// 3) FC GEMM via WMMA. M=1536, N=256, K=30752.
//    Block: 512 threads = 16 waves, owns 64x256 output tile.
//      waves 0..7  : k-half 0 (elements [0, 15360))
//      waves 8..15 : k-half 1 (elements [15360, 30752))
//    Each wave: 32x64 tile = 8 accumulators, 8 WMMA per k-step.
//    A staged to LDS in 128-element chunks via async global->LDS.
// ---------------------------------------------------------------------------
__global__ void __launch_bounds__(512)
fc_gemm_kernel(const unsigned short* __restrict__ Apool,
               const unsigned short* __restrict__ Wb,
               const float* __restrict__ bias,
               float* __restrict__ feats, unsigned short* __restrict__ featsb) {
  const int K = 30752;
  __shared__ __align__(16) unsigned char shBuf[32768];
  unsigned short* shA = (unsigned short*)shBuf;   // [2][64][128] bf16
  float* shF = (float*)shBuf;                     // combine: 4 slots x 2048 f32

  int tid  = threadIdx.x;
  int lane = tid & 31;
  int wave = tid >> 5;            // 0..15
  int kh   = wave >> 3;           // k-half
  int wl   = wave & 7;
  int mSel = wl >> 2;             // 0/1 -> rows [32*mSel, 32*mSel+32)
  int nBase = (wl & 3) << 6;      // 0,64,128,192
  int hh  = lane >> 4;
  int l15 = lane & 15;
  int blockM = blockIdx.x * 64;

  const unsigned short* bR[4];
  #pragma unroll
  for (int t = 0; t < 4; ++t) bR[t] = Wb + (size_t)(nBase + 16 * t + l15) * K;
  const unsigned short* aG0 = Apool + (size_t)(blockM + mSel * 32 + l15) * K;
  const unsigned short* aG1 = Apool + (size_t)(blockM + mSel * 32 + 16 + l15) * K;

  const unsigned short* myA0 = shA + kh * 8192 + (mSel * 32 + l15) * 128;
  const unsigned short* myA1 = myA0 + 16 * 128;

  v8f acc[8];
  #pragma unroll
  for (int i = 0; i < 8; ++i)
    #pragma unroll
    for (int r = 0; r < 8; ++r) acc[i][r] = 0.f;

  #pragma unroll 1
  for (int ci = 0; ci < 120; ++ci) {
    int kB0 = ci * 128;
    int kB1 = 15360 + ci * 128;
    // cooperative staging: both k-half A chunks, 64 rows x 128 elems each
    #pragma unroll
    for (int h = 0; h < 2; ++h) {
      int kB = h ? kB1 : kB0;
      #pragma unroll
      for (int rep = 0; rep < 2; ++rep) {
        int li  = tid + rep * 512;        // 0..1023 b128 transfers
        int row = li >> 4;
        int col = li & 15;
        stage16(Apool + (size_t)(blockM + row) * K + kB + col * 8,
                shA + h * 8192 + row * 128 + col * 8);
      }
    }
    stage_wait();
    __syncthreads();

    int kOwn = kh ? kB1 : kB0;
    __builtin_prefetch((const void*)(bR[0] + kOwn + 128), 0, 0);
    #pragma unroll
    for (int s = 0; s < 4; ++s) {
      int lk = s * 32 + hh * 8;           // element offset within LDS row
      Frag16 a0, a1;
      a0.q[0] = *(const uint4*)(myA0 + lk);
      a0.q[1] = *(const uint4*)(myA0 + lk + 16);
      a1.q[0] = *(const uint4*)(myA1 + lk);
      a1.q[1] = *(const uint4*)(myA1 + lk + 16);
      int gk = kOwn + s * 32 + hh * 8;
      #pragma unroll
      for (int t = 0; t < 4; ++t) {
        Frag16 bb;
        bb.q[0] = *(const uint4*)(bR[t] + gk);
        bb.q[1] = *(const uint4*)(bR[t] + gk + 16);
        acc[t]     = __builtin_amdgcn_wmma_f32_16x16x32_bf16(false, a0.v, false, bb.v, (short)0, acc[t],     false, false);
        acc[4 + t] = __builtin_amdgcn_wmma_f32_16x16x32_bf16(false, a1.v, false, bb.v, (short)0, acc[4 + t], false, false);
      }
    }
    __syncthreads();
  }

  // odd K-step (K=961 steps): elements [30720, 30752), kh==1 waves, from global
  if (kh == 1) {
    int gk = 30720 + hh * 8;
    Frag16 a0, a1;
    a0.q[0] = *(const uint4*)(aG0 + gk);  a0.q[1] = *(const uint4*)(aG0 + gk + 16);
    a1.q[0] = *(const uint4*)(aG1 + gk);  a1.q[1] = *(const uint4*)(aG1 + gk + 16);
    #pragma unroll
    for (int t = 0; t < 4; ++t) {
      Frag16 bb;
      bb.q[0] = *(const uint4*)(bR[t] + gk);
      bb.q[1] = *(const uint4*)(bR[t] + gk + 16);
      acc[t]     = __builtin_amdgcn_wmma_f32_16x16x32_bf16(false, a0.v, false, bb.v, (short)0, acc[t],     false, false);
      acc[4 + t] = __builtin_amdgcn_wmma_f32_16x16x32_bf16(false, a1.v, false, bb.v, (short)0, acc[4 + t], false, false);
    }
  }

  __syncthreads();
  // combine kh=1 partials into kh=0 accumulators via LDS, two 32 KB phases
  #pragma unroll 1
  for (int ph = 0; ph < 2; ++ph) {
    if (kh == 1 && (wl >> 2) == ph) {
      int slot = wl & 3;
      #pragma unroll
      for (int a = 0; a < 2; ++a)
        #pragma unroll
        for (int t = 0; t < 4; ++t)
          #pragma unroll
          for (int r = 0; r < 8; ++r)
            shF[slot * 2048 + (a * 16 + r + 8 * hh) * 64 + t * 16 + l15] = acc[a * 4 + t][r];
    }
    __syncthreads();
    if (kh == 0 && (wl >> 2) == ph) {
      int slot = wl & 3;
      #pragma unroll
      for (int a = 0; a < 2; ++a)
        #pragma unroll
        for (int t = 0; t < 4; ++t)
          #pragma unroll
          for (int r = 0; r < 8; ++r)
            acc[a * 4 + t][r] += shF[slot * 2048 + (a * 16 + r + 8 * hh) * 64 + t * 16 + l15];
    }
    __syncthreads();
  }

  // epilogue by kh==0 waves
  if (kh == 0) {
    #pragma unroll
    for (int a = 0; a < 2; ++a)
      #pragma unroll
      for (int t = 0; t < 4; ++t)
        #pragma unroll
        for (int r = 0; r < 8; ++r) {
          int m = blockM + mSel * 32 + a * 16 + r + 8 * hh;
          int n = nBase + t * 16 + l15;
          float v = fmaxf(acc[a * 4 + t][r] + bias[n], 0.f);
          feats[(size_t)m * 256 + n]  = v;
          featsb[(size_t)m * 256 + n] = f2bf(v);
        }
  }
}

// ---------------------------------------------------------------------------
// 4) row norms n2[b,s] = ||feats[b,s]||^2
// ---------------------------------------------------------------------------
__global__ void __launch_bounds__(256)
norms_kernel(const float* __restrict__ feats, float* __restrict__ n2) {
  int row = blockIdx.x * 256 + threadIdx.x;
  if (row >= 1536) return;
  const float* f = feats + (size_t)row * 256;
  float s = 0.f;
  for (int k = 0; k < 256; ++k) s = fmaf(f[k], f[k], s);
  n2[row] = s;
}

// ---------------------------------------------------------------------------
// 5) pairwise Gram via WMMA: Draw[b,i,j] = n2_i + n2_j - 2 <f_i,f_j>
// ---------------------------------------------------------------------------
__global__ void __launch_bounds__(256)
pairwise_kernel(const unsigned short* __restrict__ featsb,
                const float* __restrict__ n2, float* __restrict__ Draw) {
  int lane = threadIdx.x & 31;
  int wave = (blockIdx.x << 3) + (threadIdx.x >> 5);
  int b = wave / 36, tile = wave % 36;
  int iBase = (tile / 6) << 4, jBase = (tile % 6) << 4;
  int hh = lane >> 4, l15 = lane & 15;

  const unsigned short* aRow = featsb + (size_t)(b * 96 + iBase + l15) * 256;
  const unsigned short* bRow = featsb + (size_t)(b * 96 + jBase + l15) * 256;

  v8f acc;
  #pragma unroll
  for (int r = 0; r < 8; ++r) acc[r] = 0.f;

  #pragma unroll 1
  for (int kk = 0; kk < 256; kk += 32) {
    int gk = kk + hh * 8;
    Frag16 a, bb;
    a.q[0]  = *(const uint4*)(aRow + gk);
    a.q[1]  = *(const uint4*)(aRow + gk + 16);
    bb.q[0] = *(const uint4*)(bRow + gk);
    bb.q[1] = *(const uint4*)(bRow + gk + 16);
    acc = __builtin_amdgcn_wmma_f32_16x16x32_bf16(false, a.v, false, bb.v, (short)0, acc, false, false);
  }

  #pragma unroll
  for (int r = 0; r < 8; ++r) {
    int i = iBase + r + 8 * hh;
    int j = jBase + l15;
    Draw[(size_t)b * 9216 + i * 96 + j] = n2[b * 96 + i] + n2[b * 96 + j] - 2.f * acc[r];
  }
}

// ---------------------------------------------------------------------------
// 6) D[b,i,j] = max(Draw[b,i,j] - Draw[b,j,j], 0)
// ---------------------------------------------------------------------------
__global__ void __launch_bounds__(256)
diag_clamp_kernel(const float* __restrict__ Draw, float* __restrict__ D) {
  int idx = blockIdx.x * 256 + threadIdx.x;
  if (idx >= 16 * 9216) return;
  int b = idx / 9216, j = idx % 96;
  float v = Draw[idx] - Draw[(size_t)b * 9216 + j * 96 + j];
  D[idx] = fmaxf(v, 0.f);
}

__global__ void __launch_bounds__(256)
zero_kernel(float* __restrict__ p, int n) {
  int idx = blockIdx.x * 256 + threadIdx.x;
  if (idx < n) p[idx] = 0.f;
}

// ---------------------------------------------------------------------------
// 7) OT step part P: gamma = Q*softmax(A); p = col sums; cost = <gamma,D>.
// ---------------------------------------------------------------------------
__global__ void __launch_bounds__(96)
ot_p_kernel(const float* __restrict__ A, const float* __restrict__ Q,
            const float* __restrict__ D, float* __restrict__ p,
            float* __restrict__ cost) {
  int bk = blockIdx.x;
  int b  = bk / 10;
  int i  = threadIdx.x;
  __shared__ float pl[96];
  __shared__ float red[96];
  pl[i] = 0.f;
  __syncthreads();

  const float* Arow = A + ((size_t)bk * 96 + i) * 96;
  const float* Drow = D + ((size_t)b * 96 + i) * 96;
  float q = Q[(size_t)bk * 96 + i];

  float m = -1e30f;
  for (int j = 0; j < 96; ++j) m = fmaxf(m, Arow[j]);
  float Z = 0.f;
  for (int j = 0; j < 96; ++j) Z += __expf(Arow[j] - m);
  float inv = 1.f / Z;
  float cacc = 0.f;
  for (int j = 0; j < 96; ++j) {
    float g = q * __expf(Arow[j] - m) * inv;
    atomicAdd(&pl[j], g);                      // ds_add_f32
    cacc = fmaf(g, Drow[j], cacc);
  }
  red[i] = cacc;
  __syncthreads();
  if (i < 32) red[i] = red[i] + red[i + 32] + red[i + 64];
  __syncthreads();
  if (i == 0) {
    float t = 0.f;
    for (int x = 0; x < 32; ++x) t += red[x];
    cost[bk] = t;
  }
  p[(size_t)bk * 96 + i] = pl[i];
}

// ---------------------------------------------------------------------------
// 8) OT step part U: softmax-row Jacobian applied to G = w + lam*D.
// ---------------------------------------------------------------------------
__global__ void __launch_bounds__(96)
ot_u_kernel(float* __restrict__ A, const float* __restrict__ Q,
            const float* __restrict__ D, const float* __restrict__ p,
            const float* __restrict__ cost, const float* __restrict__ theta) {
  int bk = blockIdx.x;
  int b = bk / 10, k = bk % 10;
  int i = threadIdx.x;
  __shared__ float wl[96];

  {
    float vals[10];
    float mx = -1e30f;
    #pragma unroll
    for (int kk = 0; kk < 10; ++kk) {
      float v = p[((size_t)b * 10 + kk) * 96 + i] * 20.0f;   // 1/TAU
      vals[kk] = v; mx = fmaxf(mx, v);
    }
    float Z = 0.f;
    #pragma unroll
    for (int kk = 0; kk < 10; ++kk) Z += __expf(vals[kk] - mx);
    wl[i] = __expf(vals[k] - mx) / Z;
  }
  __syncthreads();

  float lam = 20.0f * fmaxf(cost[bk] - theta[k], 0.f);       // 2*RHO
  float q   = Q[(size_t)bk * 96 + i];
  float* Arow = A + ((size_t)bk * 96 + i) * 96;
  const float* Drow = D + ((size_t)b * 96 + i) * 96;

  float m = -1e30f;
  for (int j = 0; j < 96; ++j) m = fmaxf(m, Arow[j]);
  float Z = 0.f;
  for (int j = 0; j < 96; ++j) Z += __expf(Arow[j] - m);
  float inv = 1.f / Z;
  float dot = 0.f;
  for (int j = 0; j < 96; ++j) {
    float s = __expf(Arow[j] - m) * inv;
    dot = fmaf(s, wl[j] + lam * Drow[j], dot);
  }
  for (int j = 0; j < 96; ++j) {
    float s = __expf(Arow[j] - m) * inv;
    float G = wl[j] + lam * Drow[j];
    Arow[j] -= 0.5f * q * s * (G - dot);                     // LR
  }
}

// ---------------------------------------------------------------------------
// Host-side launcher (all launches on `stream`, graph-capture safe).
// ---------------------------------------------------------------------------
extern "C" void kernel_launch(void* const* d_in, const int* in_sizes, int n_in,
                              void* d_out, int out_size, void* d_ws, size_t ws_size,
                              hipStream_t stream) {
  const float* X      = (const float*)d_in[0];
  const float* Q      = (const float*)d_in[1];
  const float* conv_w = (const float*)d_in[2];
  const float* conv_b = (const float*)d_in[3];
  const float* fc_w   = (const float*)d_in[4];
  const float* fc_b   = (const float*)d_in[5];
  const float* theta  = (const float*)d_in[6];

  char* ws = (char*)d_ws;
  size_t off = 0;
  unsigned short* pooled = (unsigned short*)(ws + off); off += (size_t)1536 * 30752 * 2;
  unsigned short* wb     = (unsigned short*)(ws + off); off += (size_t)256 * 30752 * 2;
  float* feats           = (float*)(ws + off);          off += (size_t)1536 * 256 * 4;
  unsigned short* featsb = (unsigned short*)(ws + off); off += (size_t)1536 * 256 * 2;
  float* n2              = (float*)(ws + off);          off += (size_t)1536 * 4;
  float* Draw            = (float*)(ws + off);          off += (size_t)16 * 9216 * 4;
  float* Dm              = (float*)(ws + off);          off += (size_t)16 * 9216 * 4;
  float* A               = (float*)(ws + off);          off += (size_t)16 * 10 * 96 * 96 * 4;
  float* p               = (float*)(ws + off);          off += (size_t)16 * 10 * 96 * 4;
  float* cost            = (float*)(ws + off);          off += (size_t)160 * 4;

  conv_pool_kernel<<<184513, 256, 0, stream>>>(X, conv_w, conv_b, pooled);
  cast_w_kernel<<<30752, 256, 0, stream>>>(fc_w, wb, 256 * 30752);
  // 24 blocks x 16 waves, 64x256 tile per block, in-block split-K
  fc_gemm_kernel<<<24, 512, 0, stream>>>(pooled, wb, fc_b, feats, featsb);
  norms_kernel<<<6, 256, 0, stream>>>(feats, n2);
  pairwise_kernel<<<72, 256, 0, stream>>>(featsb, n2, Draw);
  diag_clamp_kernel<<<576, 256, 0, stream>>>(Draw, Dm);
  zero_kernel<<<5760, 256, 0, stream>>>(A, 16 * 10 * 96 * 96);
  for (int it = 0; it < 60; ++it) {
    ot_p_kernel<<<160, 96, 0, stream>>>(A, Q, Dm, p, cost);
    ot_u_kernel<<<160, 96, 0, stream>>>(A, Q, Dm, p, cost, theta);
  }
  ot_p_kernel<<<160, 96, 0, stream>>>(A, Q, Dm, (float*)d_out, cost);
}
